// Agent_75866302316690
// MI455X (gfx1250) — matrix operand order repaired and checked by reading, not compiled, not run on hardware
//
#include <hip/hip_runtime.h>
#include <hip/hip_bf16.h>

// ---------------------------------------------------------------------------
// Card-game transformer (B=1024, C=128, enc seq 81->96, 2 enc + 2 dec layers,
// 32 actions) for MI455X / gfx1250.
//   * GEMMs via v_wmma_f32_16x16x32_f16, K templated -> fully preloaded
//     fragment arrays (loads pipelined ahead of the WMMA chain)
//   * attention: per-sample block, K matrix DMA'd to LDS once via the
//     Tensor Data Mover (tensor_load_to_lds + s_wait_tensorcnt), B fragments
//     from LDS (ds_load_b128), scores+softmax fused
//   * ID-embedding projection (81920 x 1024 x 64, the biggest GEMM) via WMMA
//     with gathered rows staged f32->f16 into LDS
//   * intermediates in d_ws (~190MB) -> resident in the 192MB L2
// ---------------------------------------------------------------------------

typedef _Float16 half_t;
typedef __attribute__((ext_vector_type(16))) _Float16 v16h;
typedef __attribute__((ext_vector_type(8)))  _Float16 v8h;
typedef __attribute__((ext_vector_type(8)))  float    v8f;
typedef __attribute__((ext_vector_type(4)))  unsigned int v4u;
typedef __attribute__((ext_vector_type(8)))  int v8i;
typedef __attribute__((ext_vector_type(4)))  int v4i;

#define DEVINL __device__ __forceinline__

DEVINL v16h frag_join(v8h lo, v8h hi) {
  v16h r;
#pragma unroll
  for (int i = 0; i < 8; ++i) { r[i] = lo[i]; r[i + 8] = hi[i]; }
  return r;
}

// A fragment (16x32 f16): lanes 0-15 -> M=lane, K base 0; lanes 16-31 -> K base 8.
DEVINL v16h load_a_frag(const half_t* A, int lda, int m0, int k0, int lane) {
  int m  = m0 + (lane & 15);
  int kb = k0 + ((lane >> 4) << 3);
  const half_t* p = A + (size_t)m * lda + kb;
  v8h lo = *(const v8h*)p;
  v8h hi = *(const v8h*)(p + 16);
  return frag_join(lo, hi);
}
// B fragment (32x16 f16) from Bt stored NxK row-major: lanes 0-15 -> N=lane,
// K base 0; lanes 16-31 -> K base 8 (same k order as A fragment).
DEVINL v16h load_b_frag(const half_t* Bt, int ldb, int n0, int k0, int lane) {
  int n  = n0 + (lane & 15);
  int kb = k0 + ((lane >> 4) << 3);
  const half_t* p = Bt + (size_t)n * ldb + kb;
  v8h lo = *(const v8h*)p;
  v8h hi = *(const v8h*)(p + 16);
  return frag_join(lo, hi);
}
DEVINL v8f wmma16(v16h a, v16h b, v8f c) {
  return __builtin_amdgcn_wmma_f32_16x16x32_f16(false, a, false, b, (short)0, c,
                                                false, false);
}
DEVINL float wred32(float v) {
#pragma unroll
  for (int o = 16; o; o >>= 1) v += __shfl_xor(v, o, 32);
  return v;
}

// Tensor Data Mover: DMA n_elem f16 elements (1 row) global -> LDS.
DEVINL void tdm_load_to_lds(half_t* lds_dst, const half_t* gsrc, int n_elem) {
  unsigned lds_off = (unsigned)(size_t)lds_dst;  // low 32 bits = LDS byte offset
  unsigned long long ga = (unsigned long long)(size_t)gsrc;
  v4u g0;
  g0[0] = 1u;                                     // count=1, user mode
  g0[1] = lds_off;                                // lds_addr
  g0[2] = (unsigned)(ga & 0xffffffffu);           // global_addr[31:0]
  g0[3] = (unsigned)((ga >> 32) & 0x1ffffffu) | (2u << 30);  // addr[56:32], type=2
  v8i g1;
  g1[0] = 0x00010000;                             // data_size=2B, no multicast
  g1[1] = (int)(((unsigned)n_elem & 0xffffu) << 16);     // tensor_dim0 lo16
  g1[2] = (int)(((unsigned)n_elem >> 16) | (1u << 16));  // dim0 hi16, tensor_dim1=1
  g1[3] = (int)(((unsigned)n_elem & 0xffffu) << 16);     // tile_dim0 = n_elem
  g1[4] = 1;                                      // tile_dim1=1, tile_dim2=0
  g1[5] = n_elem;                                 // tensor_dim0_stride
  g1[6] = 0;
  g1[7] = 0;
  v4i gz = {0, 0, 0, 0};
#if defined(__clang_major__) && (__clang_major__ >= 23)
  v8i gz8 = {0, 0, 0, 0, 0, 0, 0, 0};
  __builtin_amdgcn_tensor_load_to_lds(g0, g1, gz, gz, gz8, 0);
#else
  __builtin_amdgcn_tensor_load_to_lds(g0, g1, gz, gz, 0);
#endif
  __builtin_amdgcn_s_wait_tensorcnt(0);
}

DEVINL void store_tile(v8f acc, int m0, int n0, int lane,
                       const float* bias, const half_t* Res,
                       half_t* Out, half_t* OutT, int mtps, int relu) {
  int n = n0 + (lane & 15);
  float bn = bias ? bias[n] : 0.f;
#pragma unroll
  for (int r = 0; r < 8; ++r) {
    int m = m0 + r + ((lane >> 4) << 3);
    float v = acc[r] + bn;
    if (relu) v = v > 0.f ? v : 0.f;
    if (Res) v += (float)Res[(size_t)m * 128 + n];
    if (OutT) {                      // transposed per-sample: [b][n][t]
      int rows = mtps * 16;
      int sb = m / rows, t = m % rows;
      OutT[((size_t)sb * 128 + n) * rows + t] = (half_t)v;
    } else {
      Out[(size_t)m * 128 + n] = (half_t)v;
    }
  }
}

// ---------------------------------------------------------------------------
// Batched WMMA GEMM, K compile-time: Out[m][n] = act(sum_k A[m][k]*Bt[n][k]+b)
// block = 128 (4 waves); wave handles m-tile blockIdx.x and n-tiles w, w+4
// (nTot fixed at 128).  All fragments preloaded -> loads pipelined ahead of
// the WMMA chain.  Optional ReLU / residual / per-sample B / transposed out.
// ---------------------------------------------------------------------------
template <int K>
__global__ void wmma_gemm_kernel(const half_t* __restrict__ A, int lda,
                                 const half_t* __restrict__ Bt, int ldb,
                                 long long strideB, int mtps,
                                 const float* __restrict__ bias,
                                 const half_t* __restrict__ Res,
                                 half_t* __restrict__ Out,
                                 half_t* __restrict__ OutT, int relu) {
  constexpr int NK = K / 32;
  int wave = threadIdx.x >> 5, lane = threadIdx.x & 31;
  int mt = blockIdx.x, m0 = mt * 16;
  const half_t* Bs = Bt + (strideB ? (size_t)(mt / mtps) * (size_t)strideB : 0);
  if (Res) __builtin_prefetch(Res + ((size_t)m0 + (lane & 15)) * 128, 0, 1);
  v16h af[NK], bf0[NK], bf1[NK];
#pragma unroll
  for (int i = 0; i < NK; ++i) af[i]  = load_a_frag(A, lda, m0, i * 32, lane);
#pragma unroll
  for (int i = 0; i < NK; ++i) bf0[i] = load_b_frag(Bs, ldb, wave * 16, i * 32, lane);
#pragma unroll
  for (int i = 0; i < NK; ++i) bf1[i] = load_b_frag(Bs, ldb, (wave + 4) * 16, i * 32, lane);
  v8f a0 = {}, a1 = {};
#pragma unroll
  for (int i = 0; i < NK; ++i) a0 = wmma16(af[i], bf0[i], a0);
#pragma unroll
  for (int i = 0; i < NK; ++i) a1 = wmma16(af[i], bf1[i], a1);
  store_tile(a0, m0, wave * 16, lane, bias, Res, Out, OutT, mtps, relu);
  store_tile(a1, m0, (wave + 4) * 16, lane, bias, Res, Out, OutT, mtps, relu);
}

// ---------------------------------------------------------------------------
// ID-embedding projection: out[tok][n] = sum_k id_text[id[tok]][k]*id_w[n][k]
// + id_b[n].  M=81920 tokens, K=1024, N=64.  Each block: 16 gathered rows
// staged f32->f16 into 32KB LDS, 4 waves x 1 n-tile x 32-deep WMMA chain.
// ---------------------------------------------------------------------------
__global__ void embed_id_kernel(const int* __restrict__ cards_int,
                                const float* __restrict__ id_text,
                                const half_t* __restrict__ idw16,
                                const float* __restrict__ id_b,
                                half_t* __restrict__ IDP) {
  __shared__ half_t ash[16 * 1024];  // 32KB
  int blk = blockIdx.x;              // 16 tokens per block
  int tid = threadIdx.x;             // 128
  int tk = tid >> 3, c0 = (tid & 7) * 128;
  int tok = blk * 16 + tk;
  const float* src = id_text + (size_t)cards_int[(size_t)tok * 8] * 1024 + c0;
  half_t* dst = ash + tk * 1024 + c0;
#pragma unroll 4
  for (int c = 0; c < 128; ++c) dst[c] = (half_t)src[c];
  __syncthreads();
  int wave = tid >> 5, lane = tid & 31;
  v8f acc = {};
  for (int kt = 0; kt < 32; ++kt) {
    v16h a = load_a_frag(ash, 1024, 0, kt * 32, lane);
    v16h b = load_b_frag(idw16, 1024, wave * 16, kt * 32, lane);
    acc = wmma16(a, b, acc);
  }
  int n = wave * 16 + (lane & 15);
  float bn = id_b[n];
#pragma unroll
  for (int r = 0; r < 8; ++r) {
    int m = r + ((lane >> 4) << 3);
    IDP[(size_t)(blk * 16 + m) * 64 + n] = (half_t)(acc[r] + bn);
  }
}

// ---------------------------------------------------------------------------
// Attention scores + softmax, one block per SAMPLE, one wave per 16-query
// tile.  Per-sample K matrix DMA'd once to LDS by the TDM and shared by all
// q-tiles; scores via WMMA into LDS; row softmax with key-padding mask +
// optional action (msg==0) mask.  blockDim.x = 32 * (rowsQ/16).
// ---------------------------------------------------------------------------
__global__ void score_softmax_kernel(const half_t* __restrict__ Q,
                                     const half_t* __restrict__ Kc,
                                     int rowsQ, int rowsK, int validK,
                                     float scale,
                                     const int* __restrict__ acts,
                                     half_t* __restrict__ P) {
  __shared__ half_t ksh[96 * 128];   // 24KB
  __shared__ float  s[96 * 96];      // 36KB
  int b = blockIdx.x;
  int wave = threadIdx.x >> 5, lane = threadIdx.x & 31;
  const half_t* Kb = Kc + (size_t)b * rowsK * 128;
  if (threadIdx.x < 32) tdm_load_to_lds(ksh, Kb, rowsK * 128);
  __syncthreads();

  const half_t* Qb = Q + (size_t)b * rowsQ * 128;
  v16h af[4];
#pragma unroll
  for (int i = 0; i < 4; ++i) af[i] = load_a_frag(Qb, 128, wave * 16, i * 32, lane);
  int nkt = rowsK >> 4;
  for (int kt = 0; kt < nkt; ++kt) {
    v8f acc = {};
#pragma unroll
    for (int i = 0; i < 4; ++i)
      acc = wmma16(af[i], load_b_frag(ksh, 128, kt * 16, i * 32, lane), acc);
    int col = kt * 16 + (lane & 15);
#pragma unroll
    for (int r = 0; r < 8; ++r)
      s[(wave * 16 + r + ((lane >> 4) << 3)) * 96 + col] = acc[r] * scale;
  }
  if (lane < 16) {
    int row = wave * 16 + lane;
    float* sr = s + row * 96;
    float mx = -3.4e38f;
    for (int j = 0; j < rowsK; ++j) {
      bool ok = (j < validK) &&
                (!acts || acts[((size_t)b * 32 + j) * 7 + 1] != 0);
      float v = ok ? sr[j] : -1.0e9f;
      sr[j] = v;
      mx = v > mx ? v : mx;
    }
    float sum = 0.f;
    for (int j = 0; j < rowsK; ++j) { float e = __expf(sr[j] - mx); sr[j] = e; sum += e; }
    float inv = 1.f / sum;
    half_t* Pr = P + ((size_t)b * rowsQ + row) * rowsK;
    for (int j = 0; j < rowsK; ++j) Pr[j] = (half_t)(sr[j] * inv);
  }
}

// ---------------------------------------------------------------------------
// LayerNorm over 128 features; wave per row, 4 rows per block.
// ---------------------------------------------------------------------------
__global__ void ln_kernel(const half_t* __restrict__ X,
                          const float* __restrict__ g,
                          const float* __restrict__ bta,
                          const float* __restrict__ addv, int rowsPerSample,
                          half_t* __restrict__ Y) {
  int row  = blockIdx.x * 4 + (threadIdx.x >> 5);
  int lane = threadIdx.x & 31;
  const half_t* xr = X + (size_t)row * 128;
  float x[4];
#pragma unroll
  for (int i = 0; i < 4; ++i) x[i] = (float)xr[lane * 4 + i];
  float mean = wred32(x[0] + x[1] + x[2] + x[3]) * (1.f / 128.f);
  float vs = 0.f;
#pragma unroll
  for (int i = 0; i < 4; ++i) { float d = x[i] - mean; vs += d * d; }
  float rstd = rsqrtf(wred32(vs) * (1.f / 128.f) + 1e-5f);
  half_t* yr = Y + (size_t)row * 128;
  int sb = row / rowsPerSample;
#pragma unroll
  for (int i = 0; i < 4; ++i) {
    int c = lane * 4 + i;
    float v = (x[i] - mean) * rstd;
    if (g) v = v * g[c] + bta[c];
    if (addv) v += addv[(size_t)sb * 128 + c];
    yr[c] = (half_t)v;
  }
}

// --------------------------- embedding / misc -----------------------------
__global__ void cast_kernel(const float* __restrict__ s, half_t* __restrict__ d,
                            int n) {
  int i = blockIdx.x * 256 + threadIdx.x;
  if (i < n) d[i] = (half_t)s[i];
}

__global__ void embed_kernel(const int* __restrict__ cards_int,
                             const float* __restrict__ cards_float,
                             const half_t* __restrict__ idp,
                             const float* __restrict__ owner_e,
                             const float* __restrict__ pos_e,
                             const float* __restrict__ attr_e,
                             const float* __restrict__ race_e,
                             const float* __restrict__ lvl_e,
                             const float* __restrict__ type_w,
                             const float* __restrict__ type_b,
                             const float* __restrict__ atk_w,
                             const float* __restrict__ atk_b,
                             const float* __restrict__ def_w,
                             const float* __restrict__ def_b,
                             const float* __restrict__ loc_e,
                             const float* __restrict__ seq_e,
                             half_t* __restrict__ X) {
  int tok = blockIdx.x;
  int b = tok / 80, t = tok % 80;
  const int*   ci = cards_int   + (size_t)tok * 8;
  const float* cf = cards_float + (size_t)tok * 57;
  int tid = threadIdx.x;  // 64 threads
  float acc = (float)idp[(size_t)tok * 64 + tid];  // id proj (bias applied)
  int seg = tid >> 3, j = tid & 7;
  float f = 0.f;
  if      (seg == 0) f = owner_e[ci[1] * 8 + j];
  else if (seg == 1) f = pos_e[ci[4] * 8 + j];
  else if (seg == 2) f = attr_e[ci[5] * 8 + j];
  else if (seg == 3) f = race_e[ci[6] * 8 + j];
  else if (seg == 4) f = lvl_e[ci[7] * 8 + j];
  else if (seg == 5) { f = type_b[j]; for (int k = 0; k < 25; ++k) f += cf[k]      * type_w[j * 25 + k]; }
  else if (seg == 6) { f = atk_b[j];  for (int k = 0; k < 16; ++k) f += cf[25 + k] * atk_w[j * 16 + k]; }
  else               { f = def_b[j];  for (int k = 0; k < 16; ++k) f += cf[41 + k] * def_w[j * 16 + k]; }
  __shared__ float sa[64], sb[64], st[4];
  sa[tid] = acc; sb[tid] = f;
  __syncthreads();
  if (tid == 0) {
    float m = 0, v = 0;
    for (int k = 0; k < 64; ++k) m += sa[k];
    m *= (1.f / 64.f);
    for (int k = 0; k < 64; ++k) { float d = sa[k] - m; v += d * d; }
    st[0] = m; st[1] = v * (1.f / 64.f);
  }
  if (tid == 1) {
    float m = 0, v = 0;
    for (int k = 0; k < 64; ++k) m += sb[k];
    m *= (1.f / 64.f);
    for (int k = 0; k < 64; ++k) { float d = sb[k] - m; v += d * d; }
    st[2] = m; st[3] = v * (1.f / 64.f);
  }
  __syncthreads();
  float idn = (acc - st[0]) * rsqrtf(st[1] + 1e-5f);
  float ftn = (f   - st[2]) * rsqrtf(st[3] + 1e-5f);
  int loc = ci[2], sq = ci[3];
  half_t* xr = X + ((size_t)b * 96 + t + 1) * 128;
  xr[tid]      = (half_t)(idn + loc_e[loc * 128 + tid]      + seq_e[sq * 128 + tid]);
  xr[64 + tid] = (half_t)(ftn + loc_e[loc * 128 + 64 + tid] + seq_e[sq * 128 + 64 + tid]);
}

__global__ void napad_kernel(const float* __restrict__ na, half_t* __restrict__ X) {
  int b = blockIdx.x, c = threadIdx.x;  // 128
  X[((size_t)b * 96) * 128 + c] = (half_t)na[c];
  for (int r = 81; r < 96; ++r) X[((size_t)b * 96 + r) * 128 + c] = (half_t)0.f;
}

__global__ void globalfeat_kernel(const float* __restrict__ glob,
                                  const float* __restrict__ w1,
                                  const float* __restrict__ b1,
                                  const float* __restrict__ w2,
                                  const float* __restrict__ b2,
                                  float* __restrict__ FG) {
  int b = blockIdx.x, t = threadIdx.x;  // 128
  __shared__ float h[128], ss[128], red[2];
  float a = b1[t];
  for (int k = 0; k < 44; ++k) a += glob[(size_t)b * 44 + k] * w1[t * 44 + k];
  h[t] = a > 0.f ? a : 0.f;
  __syncthreads();
  float o = b2[t];
  for (int k = 0; k < 128; ++k) o += h[k] * w2[t * 128 + k];
  ss[t] = o;
  __syncthreads();
  if (t == 0) {
    float m = 0, v = 0;
    for (int k = 0; k < 128; ++k) m += ss[k];
    m *= (1.f / 128.f);
    for (int k = 0; k < 128; ++k) { float d = ss[k] - m; v += d * d; }
    red[0] = m; red[1] = v * (1.f / 128.f);
  }
  __syncthreads();
  FG[(size_t)b * 128 + t] = (o - red[0]) * rsqrtf(red[1] + 1e-5f);
}

__global__ void actbuild_kernel(const half_t* __restrict__ FC,
                                const int* __restrict__ actions,
                                const float* __restrict__ acw,
                                const float* __restrict__ acb,
                                const float* __restrict__ msg_e,
                                const float* __restrict__ act_e,
                                const float* __restrict__ yes_e,
                                const float* __restrict__ ph_e,
                                const float* __restrict__ cn_e,
                                const float* __restrict__ pos_e,
                                half_t* __restrict__ XA) {
  int ra = blockIdx.x;           // b*32 + action
  int b  = ra >> 5;
  const int* ai = actions + (size_t)ra * 7;
  int t = threadIdx.x;           // 128
  const half_t* src = FC + ((size_t)b * 96 + ai[0]) * 128;
  if (t < 64) {
    float v = acb[t];
    for (int k = 0; k < 128; ++k) v += (float)src[k] * acw[t * 128 + k];
    XA[(size_t)ra * 128 + t] = (half_t)v;
  } else {
    int u = t - 64;
    float e2;
    if      (u < 24) e2 = msg_e[ai[1] * 24 + u];
    else if (u < 32) e2 = act_e[ai[2] * 8 + (u - 24)];
    else if (u < 40) e2 = yes_e[ai[3] * 8 + (u - 32)];
    else if (u < 48) e2 = ph_e [ai[4] * 8 + (u - 40)];
    else if (u < 56) e2 = cn_e [ai[5] * 8 + (u - 48)];
    else             e2 = pos_e[ai[6] * 8 + (u - 56)];
    XA[(size_t)ra * 128 + t] = (half_t)e2;
  }
}

__global__ void value_kernel(const half_t* __restrict__ YA,
                             const float* __restrict__ vw,
                             const float* __restrict__ vb,
                             float* __restrict__ out, int n) {
  int i = blockIdx.x * 128 + threadIdx.x;
  if (i < n) {
    const half_t* r = YA + (size_t)i * 128;
    float a = vb[0];
    for (int k = 0; k < 128; ++k) a += (float)r[k] * vw[k];
    out[i] = a;
  }
}

// ---------------------------------------------------------------------------
template <int K>
static void launch_gemm(hipStream_t st, const half_t* A, int lda,
                        const half_t* Bt, int ldb, long long strideB, int mtps,
                        const float* bias, const half_t* Res, half_t* Out,
                        half_t* OutT, int relu, int mTiles) {
  wmma_gemm_kernel<K><<<dim3((unsigned)mTiles), 128, 0, st>>>(
      A, lda, Bt, ldb, strideB, mtps, bias, Res, Out, OutT, relu);
}

extern "C" void kernel_launch(void* const* d_in, const int* in_sizes, int n_in,
                              void* d_out, int out_size, void* d_ws,
                              size_t ws_size, hipStream_t stream) {
  (void)in_sizes; (void)n_in; (void)out_size; (void)ws_size;
  const int*   cards_int   = (const int*)d_in[0];
  const float* cards_float = (const float*)d_in[1];
  const float* glob        = (const float*)d_in[2];
  const int*   actions     = (const int*)d_in[3];
  auto F = [&](int i) { return (const float*)d_in[i]; };

  enum { P_LOC = 4, P_SEQ, P_IDTEXT, P_IDW, P_IDB, P_OWNER, P_POS, P_ATTR,
         P_RACE, P_LEVEL, P_TYPEW, P_TYPEB, P_ATKW, P_ATKB, P_DEFW, P_DEFB,
         P_NA, P_GW1, P_GB1, P_GW2, P_GB2, P_AMSG, P_AACT, P_AYES, P_APH,
         P_ACN, P_APOS, P_ACW, P_ACB, P_VW, P_VB, P_ENC0 = 35, P_DEC0 = 67 };

  half_t* ws = (half_t*)d_ws;
  const size_t SZX = (size_t)1024 * 96 * 128;
  const size_t SZA = (size_t)1024 * 32 * 128;
  half_t* X  = ws;
  half_t* Yb = ws + 1 * SZX;   // LN out / later f_cards (FC)
  half_t* Qb = ws + 2 * SZX;   // Q / decoder XA,YA,QA
  half_t* Kb = ws + 3 * SZX;   // K (enc + cross)
  half_t* VT = ws + 4 * SZX;   // V^T (enc + cross)
  half_t* Ob = ws + 5 * SZX;   // attn out / FFN chunk / decoder small bufs
  half_t* Pb = ws + 6 * SZX;   // softmax probs
  float*  FG = (float*)(ws + 7 * SZX);
  half_t* Warena = ws + 7 * SZX + 262144;           // f16 weight arena (<1.2M)
  half_t* IDP    = Warena + 1310720;                // id projection, 81920x64
  half_t* XA = Qb;          half_t* YA = Qb + SZA;   half_t* QA  = Qb + 2 * SZA;
  half_t* KAs = Ob;         half_t* VTAs = Ob + SZA; half_t* OA  = Ob + 2 * SZA;
  half_t* PAs = Pb;         half_t* PAx = Pb + SZA;

  size_t wcur = 0;
  auto castW = [&](int idx, size_t n) -> half_t* {
    half_t* dst = Warena + wcur;
    wcur += (n + 255) & ~(size_t)255;
    cast_kernel<<<dim3((unsigned)((n + 255) / 256)), 256, 0, stream>>>(F(idx), dst, (int)n);
    return dst;
  };
  half_t* idw16 = castW(P_IDW, 64 * 1024);
  half_t *eWq[2], *eWk[2], *eWv[2], *eWo[2], *eW1[2], *eW2[2];
  for (int L = 0; L < 2; ++L) {
    int bse = P_ENC0 + 16 * L;
    eWq[L] = castW(bse + 8, 16384);  eWk[L] = castW(bse + 10, 16384);
    eWv[L] = castW(bse + 12, 16384); eWo[L] = castW(bse + 14, 16384);
    eW1[L] = castW(bse + 4, 512 * 128); eW2[L] = castW(bse + 6, 128 * 512);
  }
  half_t *dSWq[2], *dSWk[2], *dSWv[2], *dSWo[2];
  half_t *dCWq[2], *dCWk[2], *dCWv[2], *dCWo[2], *dW1[2], *dW2[2];
  for (int L = 0; L < 2; ++L) {
    int bse = P_DEC0 + 26 * L;
    dSWq[L] = castW(bse + 10, 16384); dSWk[L] = castW(bse + 12, 16384);
    dSWv[L] = castW(bse + 14, 16384); dSWo[L] = castW(bse + 16, 16384);
    dCWq[L] = castW(bse + 18, 16384); dCWk[L] = castW(bse + 20, 16384);
    dCWv[L] = castW(bse + 22, 16384); dCWo[L] = castW(bse + 24, 16384);
    dW1[L]  = castW(bse + 6, 512 * 128); dW2[L] = castW(bse + 8, 128 * 512);
  }

  const float SCALE = 0.08838834764831845f;  // 1/sqrt(128)
  const int mtE = 1024 * 6, mtD = 1024 * 2;

  // ---- input embedding (id projection on WMMA, rest fused) ----
  embed_id_kernel<<<1024 * 80 / 16, 128, 0, stream>>>(cards_int, F(P_IDTEXT),
                                                      idw16, F(P_IDB), IDP);
  embed_kernel<<<1024 * 80, 64, 0, stream>>>(
      cards_int, cards_float, IDP, F(P_OWNER), F(P_POS), F(P_ATTR), F(P_RACE),
      F(P_LEVEL), F(P_TYPEW), F(P_TYPEB), F(P_ATKW), F(P_ATKB), F(P_DEFW),
      F(P_DEFB), F(P_LOC), F(P_SEQ), X);
  napad_kernel<<<1024, 128, 0, stream>>>(F(P_NA), X);
  globalfeat_kernel<<<1024, 128, 0, stream>>>(glob, F(P_GW1), F(P_GB1),
                                              F(P_GW2), F(P_GB2), FG);

  // ---- encoder layers ----
  for (int L = 0; L < 2; ++L) {
    int bse = P_ENC0 + 16 * L;
    ln_kernel<<<1024 * 96 / 4, 128, 0, stream>>>(X, F(bse + 0), F(bse + 1), nullptr, 96, Yb);
    launch_gemm<128>(stream, Yb, 128, eWq[L], 128, 0, 6, F(bse + 9),  nullptr, Qb, nullptr, 0, mtE);
    launch_gemm<128>(stream, Yb, 128, eWk[L], 128, 0, 6, F(bse + 11), nullptr, Kb, nullptr, 0, mtE);
    launch_gemm<128>(stream, Yb, 128, eWv[L], 128, 0, 6, F(bse + 13), nullptr, nullptr, VT, 0, mtE);
    score_softmax_kernel<<<1024, 192, 0, stream>>>(Qb, Kb, 96, 96, 81, SCALE, nullptr, Pb);
    launch_gemm<96>(stream, Pb, 96, VT, 96, (long long)128 * 96, 6, nullptr, nullptr, Ob, nullptr, 0, mtE);
    launch_gemm<128>(stream, Ob, 128, eWo[L], 128, 0, 6, F(bse + 15), X, X, nullptr, 0, mtE);
    ln_kernel<<<1024 * 96 / 4, 128, 0, stream>>>(X, F(bse + 2), F(bse + 3), nullptr, 96, Yb);
    for (int j = 0; j < 4; ++j) {  // FFN 512 hidden, chunked 4x128
      launch_gemm<128>(stream, Yb, 128, eW1[L] + (size_t)j * 128 * 128, 128, 0, 6,
                       F(bse + 5) + j * 128, nullptr, Ob, nullptr, 1, mtE);
      launch_gemm<128>(stream, Ob, 128, eW2[L] + (size_t)j * 128, 512, 0, 6,
                       (j == 0) ? F(bse + 7) : nullptr, X, X, nullptr, 0, mtE);
    }
  }
  ln_kernel<<<1024 * 96 / 4, 128, 0, stream>>>(X, nullptr, nullptr, FG, 96, Yb);

  actbuild_kernel<<<1024 * 32, 128, 0, stream>>>(
      Yb, actions, F(P_ACW), F(P_ACB), F(P_AMSG), F(P_AACT), F(P_AYES),
      F(P_APH), F(P_ACN), F(P_APOS), XA);

  // ---- decoder layers ----
  for (int L = 0; L < 2; ++L) {
    int bse = P_DEC0 + 26 * L;
    ln_kernel<<<1024 * 32 / 4, 128, 0, stream>>>(XA, F(bse + 0), F(bse + 1), nullptr, 32, YA);
    launch_gemm<128>(stream, YA, 128, dSWq[L], 128, 0, 2, F(bse + 11), nullptr, QA, nullptr, 0, mtD);
    launch_gemm<128>(stream, YA, 128, dSWk[L], 128, 0, 2, F(bse + 13), nullptr, KAs, nullptr, 0, mtD);
    launch_gemm<128>(stream, YA, 128, dSWv[L], 128, 0, 2, F(bse + 15), nullptr, nullptr, VTAs, 0, mtD);
    score_softmax_kernel<<<1024, 64, 0, stream>>>(QA, KAs, 32, 32, 32, SCALE, actions, PAs);
    launch_gemm<32>(stream, PAs, 32, VTAs, 32, (long long)128 * 32, 2, nullptr, nullptr, OA, nullptr, 0, mtD);
    launch_gemm<128>(stream, OA, 128, dSWo[L], 128, 0, 2, F(bse + 17), XA, XA, nullptr, 0, mtD);
    ln_kernel<<<1024 * 32 / 4, 128, 0, stream>>>(XA, F(bse + 2), F(bse + 3), nullptr, 32, YA);
    launch_gemm<128>(stream, YA, 128, dCWq[L], 128, 0, 2, F(bse + 19), nullptr, QA, nullptr, 0, mtD);
    launch_gemm<128>(stream, Yb, 128, dCWk[L], 128, 0, 6, F(bse + 21), nullptr, Kb, nullptr, 0, mtE);
    launch_gemm<128>(stream, Yb, 128, dCWv[L], 128, 0, 6, F(bse + 23), nullptr, nullptr, VT, 0, mtE);
    score_softmax_kernel<<<1024, 64, 0, stream>>>(QA, Kb, 32, 96, 81, SCALE, nullptr, PAx);
    launch_gemm<96>(stream, PAx, 96, VT, 96, (long long)128 * 96, 2, nullptr, nullptr, OA, nullptr, 0, mtD);
    launch_gemm<128>(stream, OA, 128, dCWo[L], 128, 0, 2, F(bse + 25), XA, XA, nullptr, 0, mtD);
    ln_kernel<<<1024 * 32 / 4, 128, 0, stream>>>(XA, F(bse + 4), F(bse + 5), nullptr, 32, YA);
    for (int j = 0; j < 4; ++j) {
      launch_gemm<128>(stream, YA, 128, dW1[L] + (size_t)j * 128 * 128, 128, 0, 2,
                       F(bse + 7) + j * 128, nullptr, OA, nullptr, 1, mtD);
      launch_gemm<128>(stream, OA, 128, dW2[L] + (size_t)j * 128, 512, 0, 2,
                       (j == 0) ? F(bse + 9) : nullptr, XA, XA, nullptr, 0, mtD);
    }
  }

  ln_kernel<<<1024 * 32 / 4, 128, 0, stream>>>(XA, nullptr, nullptr, nullptr, 32, YA);
  value_kernel<<<(32768 + 127) / 128, 128, 0, stream>>>(YA, F(P_VW), F(P_VB),
                                                        (float*)d_out, 32768);
}
